// GNSMsg_EdgeSelfAttn_87041807221083
// MI455X (gfx1250) — compile-verified, matrix operand order checked by reading, not run on hardware
//
#include <hip/hip_runtime.h>
#include <hip/hip_bf16.h>
#include <math.h>

typedef __attribute__((ext_vector_type(16))) _Float16 v16h;
typedef __attribute__((ext_vector_type(8)))  _Float16 v8h;
typedef __attribute__((ext_vector_type(8)))  float    v8f;

// Async global->LDS builtin takes pointers to GCC-vector int4 with explicit
// address spaces (param types confirmed by the round-2 diagnostic).
typedef int v4i_vec __attribute__((vector_size(16)));
typedef __attribute__((address_space(1))) v4i_vec as1_v4i;
typedef __attribute__((address_space(3))) v4i_vec as3_v4i;

#if defined(__gfx1250__) && __has_builtin(__builtin_amdgcn_global_load_async_to_lds_b128)
#define USE_ASYNC_LDS 1
#else
#define USE_ASYNC_LDS 0
#endif

// ---------------------------------------------------------------------------
// Utility kernels
// ---------------------------------------------------------------------------
__global__ __launch_bounds__(256) void zero_u32_kernel(unsigned* __restrict__ p, int n) {
    int i = blockIdx.x * 256 + threadIdx.x;
    if (i < n) p[i] = 0u;
}

__global__ __launch_bounds__(256) void cvt_f32_f16_kernel(const float* __restrict__ in,
                                                          _Float16* __restrict__ out, int n) {
    int i = blockIdx.x * 256 + threadIdx.x;
    if (i < n) out[i] = (_Float16)in[i];
}

// Transpose + convert all weight matrices to f16 [Dout][K] layout so the WMMA
// B-operand fragment (column of B == row of Wt) is two contiguous 16B loads.
__global__ __launch_bounds__(256) void prep_weights_kernel(
    const float* __restrict__ Wq, const float* __restrict__ Wk, const float* __restrict__ Wv,
    const float* __restrict__ Wo, const float* __restrict__ Wf1, const float* __restrict__ Wf2,
    _Float16* __restrict__ qkvT, _Float16* __restrict__ woT,
    _Float16* __restrict__ f1T,  _Float16* __restrict__ f2T) {
    int i = blockIdx.x * 256 + threadIdx.x;
    if (i < 12288) {                       // qkvT: [192][64], n<64:Q, <128:K, else V
        int n = i / 64, k = i % 64;
        const float* W = (n < 64) ? Wq : ((n < 128) ? Wk : Wv);
        qkvT[i] = (_Float16)W[k * 64 + (n & 63)];
    } else if (i < 12288 + 4096) {         // woT: [64][64]
        int j = i - 12288; int n = j / 64, k = j % 64;
        woT[j] = (_Float16)Wo[k * 64 + n];
    } else if (i < 12288 + 4096 + 16384) { // f1T: [256][64], Wf1 is [64][256]
        int j = i - 12288 - 4096; int n = j / 64, k = j % 64;
        f1T[j] = (_Float16)Wf1[k * 256 + n];
    } else if (i < 12288 + 4096 + 16384 + 16384) { // f2T: [64][256], Wf2 is [256][64]
        int j = i - 12288 - 4096 - 16384; int n = j / 256, k = j % 256;
        f2T[j] = (_Float16)Wf2[k * 64 + n];
    }
}

// ---------------------------------------------------------------------------
// LayerNorm: one wave (32 lanes) per 64-element row, shuffle reduction.
// Rows >= N (padding) are written as zeros so WMMA A-tiles stay clean.
// ---------------------------------------------------------------------------
__global__ __launch_bounds__(256) void ln_kernel(const float* __restrict__ x,
                                                 const float* __restrict__ g,
                                                 const float* __restrict__ b,
                                                 _Float16* __restrict__ y, int N) {
    int lane = threadIdx.x & 31;
    int row  = blockIdx.x * 8 + (threadIdx.x >> 5);
    _Float16* yr = y + (size_t)row * 64;
    if (row >= N) { yr[lane] = (_Float16)0.0f; yr[lane + 32] = (_Float16)0.0f; return; }
    const float* xr = x + (size_t)row * 64;
    float v0 = xr[lane], v1 = xr[lane + 32];
    float s = v0 + v1, q = v0 * v0 + v1 * v1;
#pragma unroll
    for (int m = 16; m > 0; m >>= 1) {
        s += __shfl_xor(s, m, 32);
        q += __shfl_xor(q, m, 32);
    }
    float mean = s * (1.0f / 64.0f);
    float var  = q * (1.0f / 64.0f) - mean * mean;
    float inv  = rsqrtf(var + 1e-5f);
    yr[lane]      = (_Float16)((v0 - mean) * inv * g[lane]      + b[lane]);
    yr[lane + 32] = (_Float16)((v1 - mean) * inv * g[lane + 32] + b[lane + 32]);
}

// ---------------------------------------------------------------------------
// Fragment loaders for the software-pipelined WMMA loop.
// ---------------------------------------------------------------------------
__device__ __forceinline__ v16h load_a_frag(const _Float16* arow, int k0) {
    v8h lo = *(const v8h*)(arow + k0);
    v8h hi = *(const v8h*)(arow + k0 + 16);
    v16h a;
#pragma unroll
    for (int i = 0; i < 8; i++) { a[i] = lo[i]; a[i + 8] = hi[i]; }
    return a;
}
__device__ __forceinline__ v16h load_b_frag(const _Float16* bbase, int K, int t, int k0) {
    const _Float16* p = bbase + t * 16 * K + k0;
    v8h lo = *(const v8h*)p;
    v8h hi = *(const v8h*)(p + 16);
    v16h b;
#pragma unroll
    for (int i = 0; i < 8; i++) { b[i] = lo[i]; b[i + 8] = hi[i]; }
    return b;
}

// ---------------------------------------------------------------------------
// WMMA GEMM: out[N x Dout] = A[N x K](f16) * Wt[Dout x K]^T(f16) (+bias)(+gelu)
// (+residual). Block = 256 thr = 8 waves; each wave computes 16 rows x 64 cols
// with 4 accumulator tiles, stepping K by 32 (v_wmma_f32_16x16x32_f16).
//
// The 64-col x K weight panel is shared by all 8 waves: stage it once in LDS
// (async global->LDS copy engine, ASYNCcnt-tracked), then read B fragments
// with ds_load_b128. The k-loop is 1-deep software pipelined: fragments for
// k0+32 are fetched before the four WMMAs for k0 issue, so loads run in the
// shadow of the matrix pipe.
// ---------------------------------------------------------------------------
__global__ __launch_bounds__(256) void gemm16_kernel(
    const _Float16* __restrict__ A, int lda,
    const _Float16* __restrict__ Wt, int K,
    const float* __restrict__ bias,      // [DoutTotal] or nullptr
    const float* __restrict__ resid,     // [rows][ldo] or nullptr
    float* __restrict__ outF,            // f32 output (or nullptr)
    _Float16* __restrict__ outH,         // f16 output (used when outF==nullptr)
    int ldo, int Nrows, int act) {
    __shared__ __align__(16) _Float16 Bsh[64 * 256];   // 32 KB: max panel (K=256)

    const int lane = threadIdx.x & 31;
    const int wave = threadIdx.x >> 5;
    const int row0 = blockIdx.x * 128 + wave * 16;
    const int col0 = blockIdx.y * 64;
    const int hi   = lane >> 4;
    const int r    = lane & 15;

    // --- Stage B panel: Bsh[n][k] = Wt[col0+n][k], n in [0,64), k in [0,K) ---
    const int chunks = (64 * K) / 8;  // 16-byte chunks; K is a multiple of 32
    for (int c = threadIdx.x; c < chunks; c += 256) {
        const _Float16* g = Wt + (size_t)col0 * K + c * 8;  // panel is contiguous
#if USE_ASYNC_LDS
        __builtin_amdgcn_global_load_async_to_lds_b128(
            (as1_v4i*)g, (as3_v4i*)&Bsh[c * 8], 0, 0);
#else
        *(v8h*)&Bsh[c * 8] = *(const v8h*)g;
#endif
    }
#if USE_ASYNC_LDS
#if __has_builtin(__builtin_amdgcn_s_wait_asynccnt)
    __builtin_amdgcn_s_wait_asynccnt(0);
#else
    asm volatile("s_wait_asynccnt 0x0" ::: "memory");
#endif
#endif
    __syncthreads();

    v8f c0 = {}, c1 = {}, c2 = {}, c3 = {};

    const _Float16* arow  = A + (size_t)(row0 + r) * lda + hi * 8;
    const _Float16* bbase = &Bsh[r * K + hi * 8];

    // Prologue: fragments for k0 = 0.
    v16h a  = load_a_frag(arow, 0);
    v16h b0 = load_b_frag(bbase, K, 0, 0);
    v16h b1 = load_b_frag(bbase, K, 1, 0);
    v16h b2 = load_b_frag(bbase, K, 2, 0);
    v16h b3 = load_b_frag(bbase, K, 3, 0);

    for (int k0 = 32; k0 < K; k0 += 32) {
        // Fetch next step's fragments before consuming the current ones.
        v16h na  = load_a_frag(arow, k0);
        v16h nb0 = load_b_frag(bbase, K, 0, k0);
        v16h nb1 = load_b_frag(bbase, K, 1, k0);
        v16h nb2 = load_b_frag(bbase, K, 2, k0);
        v16h nb3 = load_b_frag(bbase, K, 3, k0);
        c0 = __builtin_amdgcn_wmma_f32_16x16x32_f16(false, a, false, b0, (short)0, c0, false, false);
        c1 = __builtin_amdgcn_wmma_f32_16x16x32_f16(false, a, false, b1, (short)0, c1, false, false);
        c2 = __builtin_amdgcn_wmma_f32_16x16x32_f16(false, a, false, b2, (short)0, c2, false, false);
        c3 = __builtin_amdgcn_wmma_f32_16x16x32_f16(false, a, false, b3, (short)0, c3, false, false);
        a = na; b0 = nb0; b1 = nb1; b2 = nb2; b3 = nb3;
    }
    c0 = __builtin_amdgcn_wmma_f32_16x16x32_f16(false, a, false, b0, (short)0, c0, false, false);
    c1 = __builtin_amdgcn_wmma_f32_16x16x32_f16(false, a, false, b1, (short)0, c1, false, false);
    c2 = __builtin_amdgcn_wmma_f32_16x16x32_f16(false, a, false, b2, (short)0, c2, false, false);
    c3 = __builtin_amdgcn_wmma_f32_16x16x32_f16(false, a, false, b3, (short)0, c3, false, false);

    // Epilogue: C/D layout -> VGPR v, lane: M = v + 8*(lane>>4), N = lane&15
    v8f cacc[4] = {c0, c1, c2, c3};
#pragma unroll
    for (int t = 0; t < 4; t++) {
        int col = col0 + t * 16 + r;
        float bv = bias ? bias[col] : 0.0f;
#pragma unroll
        for (int v = 0; v < 8; v++) {
            int row = row0 + v + hi * 8;
            if (row < Nrows) {
                float val = cacc[t][v] + bv;
                if (act) val = 0.5f * val * (1.0f + erff(val * 0.70710678118654752f));
                if (resid) val += resid[(size_t)row * ldo + col];
                if (outF) outF[(size_t)row * ldo + col] = val;
                else      outH[(size_t)row * ldo + col] = (_Float16)val;
            }
        }
    }
}

// ---------------------------------------------------------------------------
// Edge-bias MLP: bias[e][h] = leaky_relu(ef@We1+be1, 0.1) @ We2 + be2
// ---------------------------------------------------------------------------
__global__ __launch_bounds__(256) void edge_bias_kernel(
    const float* __restrict__ ef, const float* __restrict__ We1, const float* __restrict__ be1,
    const float* __restrict__ We2, const float* __restrict__ be2,
    float* __restrict__ out, int E) {
    int e = blockIdx.x * 256 + threadIdx.x;
    if (e >= E) return;
    __builtin_prefetch(&ef[(size_t)e * 9 + 9 * 4096], 0, 0);
    float f[9];
#pragma unroll
    for (int i = 0; i < 9; i++) f[i] = ef[(size_t)e * 9 + i];
    float a0 = be2[0], a1 = be2[1], a2 = be2[2], a3 = be2[3];
#pragma unroll
    for (int hd = 0; hd < 18; hd++) {
        float h = be1[hd];
#pragma unroll
        for (int i = 0; i < 9; i++) h += f[i] * We1[i * 18 + hd];
        h = (h > 0.0f) ? h : 0.1f * h;
        a0 += h * We2[hd * 4 + 0];
        a1 += h * We2[hd * 4 + 1];
        a2 += h * We2[hd * 4 + 2];
        a3 += h * We2[hd * 4 + 3];
    }
    out[(size_t)e * 4 + 0] = a0;
    out[(size_t)e * 4 + 1] = a1;
    out[(size_t)e * 4 + 2] = a2;
    out[(size_t)e * 4 + 3] = a3;
}

// Monotonic float<->uint encoding for atomicMax-based segment max.
__device__ __forceinline__ unsigned fenc(float f) {
    unsigned u = __float_as_uint(f);
    return (u & 0x80000000u) ? ~u : (u | 0x80000000u);
}
__device__ __forceinline__ float fdec(unsigned u) {
    unsigned v = (u & 0x80000000u) ? (u & 0x7FFFFFFFu) : ~u;
    return __uint_as_float(v);
}

// logits[e][h] = dot(Q[dst,h], K[src,h])/4 + bias[e][h] (in-place over bias);
// segment max into mxenc via atomicMax on encoded value.
__global__ __launch_bounds__(256) void logits_max_kernel(
    const int* __restrict__ ei, const float* __restrict__ QKV,
    float* __restrict__ elog, unsigned* __restrict__ mxenc, int E) {
    int i = blockIdx.x * 256 + threadIdx.x;
    if (i >= E * 4) return;
    int e = i >> 2, hh = i & 3;
    __builtin_prefetch(&ei[2 * e + 16384], 0, 0);
    int src = ei[2 * e], dst = ei[2 * e + 1];
    const float4* q = (const float4*)(QKV + (size_t)dst * 192 + hh * 16);
    const float4* k = (const float4*)(QKV + (size_t)src * 192 + 64 + hh * 16);
    float d = 0.0f;
#pragma unroll
    for (int u = 0; u < 4; u++) {
        float4 a = q[u], b = k[u];
        d += a.x * b.x + a.y * b.y + a.z * b.z + a.w * b.w;
    }
    float lg = d * 0.25f + elog[i];
    elog[i] = lg;
    atomicMax(&mxenc[(size_t)dst * 4 + hh], fenc(lg));
}

__global__ __launch_bounds__(256) void denom_kernel(
    const int* __restrict__ ei, const float* __restrict__ elog,
    const unsigned* __restrict__ mxenc, float* __restrict__ den, int E) {
    int i = blockIdx.x * 256 + threadIdx.x;
    if (i >= E * 4) return;
    int e = i >> 2, hh = i & 3;
    int dst = ei[2 * e + 1];
    float mx = fdec(mxenc[(size_t)dst * 4 + hh]);
    atomicAdd(&den[(size_t)dst * 4 + hh], expf(elog[i] - mx));
}

__global__ __launch_bounds__(256) void scatter_kernel(
    const int* __restrict__ ei, const float* __restrict__ elog,
    const unsigned* __restrict__ mxenc, const float* __restrict__ den,
    const float* __restrict__ QKV, float* __restrict__ agg, int E) {
    int i = blockIdx.x * 256 + threadIdx.x;
    if (i >= E * 4) return;
    int e = i >> 2, hh = i & 3;
    __builtin_prefetch(&elog[i + 8192], 0, 0);
    int src = ei[2 * e], dst = ei[2 * e + 1];
    float mx = fdec(mxenc[(size_t)dst * 4 + hh]);
    float alpha = expf(elog[i] - mx) / (den[(size_t)dst * 4 + hh] + 1e-12f);
    const float4* v = (const float4*)(QKV + (size_t)src * 192 + 128 + hh * 16);
    float* a = agg + (size_t)dst * 64 + hh * 16;
#pragma unroll
    for (int u = 0; u < 4; u++) {
        float4 vv = v[u];
        atomicAdd(a + 4 * u + 0, alpha * vv.x);
        atomicAdd(a + 4 * u + 1, alpha * vv.y);
        atomicAdd(a + 4 * u + 2, alpha * vv.z);
        atomicAdd(a + 4 * u + 3, alpha * vv.w);
    }
}

// ---------------------------------------------------------------------------
// Host launcher
// ---------------------------------------------------------------------------
extern "C" void kernel_launch(void* const* d_in, const int* in_sizes, int n_in,
                              void* d_out, int out_size, void* d_ws, size_t ws_size,
                              hipStream_t stream) {
    (void)n_in; (void)out_size; (void)ws_size;
    const float* x   = (const float*)d_in[0];
    const int*   ei  = (const int*)  d_in[1];
    const float* ef  = (const float*)d_in[2];
    const float* Wq  = (const float*)d_in[3];
    const float* Wk  = (const float*)d_in[4];
    const float* Wv  = (const float*)d_in[5];
    const float* Wo  = (const float*)d_in[6];
    const float* We1 = (const float*)d_in[7];
    const float* be1 = (const float*)d_in[8];
    const float* We2 = (const float*)d_in[9];
    const float* be2 = (const float*)d_in[10];
    const float* g1  = (const float*)d_in[11];
    const float* b1  = (const float*)d_in[12];
    const float* g2  = (const float*)d_in[13];
    const float* b2  = (const float*)d_in[14];
    const float* Wf1 = (const float*)d_in[15];
    const float* bf1 = (const float*)d_in[16];
    const float* Wf2 = (const float*)d_in[17];
    const float* bf2 = (const float*)d_in[18];

    const int N    = in_sizes[0] / 64;
    const int E    = in_sizes[1] / 2;
    const int Npad = ((N + 127) / 128) * 128;

    char* ws = (char*)d_ws;
    size_t off = 0;
    auto alloc = [&](size_t bytes) -> char* {
        char* p = ws + off;
        off = (off + bytes + 255) & ~(size_t)255;
        return p;
    };
    _Float16* qkvT  = (_Float16*)alloc(192 * 64 * 2);
    _Float16* woT   = (_Float16*)alloc(64 * 64 * 2);
    _Float16* f1T   = (_Float16*)alloc(256 * 64 * 2);
    _Float16* f2T   = (_Float16*)alloc(64 * 256 * 2);
    _Float16* ybuf  = (_Float16*)alloc((size_t)Npad * 64 * 2);   // reused: y1 / agg16 / y2
    float*    QKV   = (float*)   alloc((size_t)Npad * 192 * 4);
    float*    elog  = (float*)   alloc((size_t)E * 4 * 4);       // bias then logits (in place)
    unsigned* mxenc = (unsigned*)alloc((size_t)Npad * 4 * 4);    // contiguous with den, agg
    float*    den   = (float*)   alloc((size_t)Npad * 4 * 4);
    float*    agg   = (float*)   alloc((size_t)Npad * 64 * 4);
    float*    x1    = (float*)   alloc((size_t)Npad * 64 * 4);
    _Float16* h16   = (_Float16*)alloc((size_t)Npad * 256 * 2);

    const int rowsBlocks = Npad / 128;
    const int eh = E * 4;

    // Zero mxenc + den + agg in one pass (contiguous, 256-aligned sizes).
    int zwords = Npad * (4 + 4 + 64);
    zero_u32_kernel<<<(zwords + 255) / 256, 256, 0, stream>>>(mxenc, zwords);

    prep_weights_kernel<<<192, 256, 0, stream>>>(Wq, Wk, Wv, Wo, Wf1, Wf2,
                                                 qkvT, woT, f1T, f2T);

    // LN1 -> y (f16), then QKV = y @ [Wq|Wk|Wv]  (WMMA)
    ln_kernel<<<Npad / 8, 256, 0, stream>>>(x, g1, b1, ybuf, N);
    gemm16_kernel<<<dim3(rowsBlocks, 3), 256, 0, stream>>>(
        ybuf, 64, qkvT, 64, nullptr, nullptr, QKV, nullptr, 192, N, 0);

    // Edge phase: bias MLP -> logits + segment max -> denom -> scatter alpha*V
    edge_bias_kernel<<<(E + 255) / 256, 256, 0, stream>>>(ef, We1, be1, We2, be2, elog, E);
    logits_max_kernel<<<(eh + 255) / 256, 256, 0, stream>>>(ei, QKV, elog, mxenc, E);
    denom_kernel<<<(eh + 255) / 256, 256, 0, stream>>>(ei, elog, mxenc, den, E);
    scatter_kernel<<<(eh + 255) / 256, 256, 0, stream>>>(ei, elog, mxenc, den, QKV, agg, E);

    // x1 = x + agg @ Wo  (WMMA, residual in epilogue)
    cvt_f32_f16_kernel<<<((Npad * 64) + 255) / 256, 256, 0, stream>>>(agg, ybuf, Npad * 64);
    gemm16_kernel<<<dim3(rowsBlocks, 1), 256, 0, stream>>>(
        ybuf, 64, woT, 64, nullptr, x, x1, nullptr, 64, N, 0);

    // FFN: LN2 -> h = gelu(y2@Wf1+bf1) (f16) -> out = x1 + h@Wf2+bf2
    ln_kernel<<<Npad / 8, 256, 0, stream>>>(x1, g2, b2, ybuf, N);
    gemm16_kernel<<<dim3(rowsBlocks, 4), 256, 0, stream>>>(
        ybuf, 64, f1T, 64, bf1, nullptr, nullptr, h16, 256, N, 1);
    gemm16_kernel<<<dim3(rowsBlocks, 1), 256, 0, stream>>>(
        h16, 256, f2T, 256, bf2, x1, (float*)d_out, nullptr, 64, N, 0);
}
